// LSTM_31404800868489
// MI455X (gfx1250) — matrix operand order repaired and checked by reading, not compiled, not run on hardware
//
#include <hip/hip_runtime.h>

#define T_ 152
#define B_ 128
#define I_ 75
#define H_ 256
#define G4_ 1024
#define L_ 16
#define C_ 625
#define EPSF 1e-5f

typedef __attribute__((ext_vector_type(16))) __bf16 v16bf;
typedef __attribute__((ext_vector_type(8)))  float  v8f;

union FragBF { v16bf v; uint4 q[2]; };

__device__ __forceinline__ unsigned short f2bf(float f) {
  union { float f; unsigned u; } x; x.f = f;
  unsigned r = x.u + 0x7FFFu + ((x.u >> 16) & 1u);
  return (unsigned short)(r >> 16);
}
__device__ __forceinline__ unsigned pack2bf(float a, float b) {
  return (unsigned)f2bf(a) | ((unsigned)f2bf(b) << 16);
}
__device__ __forceinline__ float sigm(float x) { return 1.0f / (1.0f + __expf(-x)); }

// ---------------------------------------------------------------------------
// Phase A: wi[t*B+b, :] = x[t,b,:] @ W  (bf16 WMMA, fp32 accum)
// block = 256 threads (8 waves), tile 128(M) x 64(N), K staged 32 at a time.
// wave w owns M-tile w (rows w*16..w*16+15) and all 4 N-tiles.
// LAYER0=1: A element comes from sequences (B,T,I) with transpose indexing and
// K=75 bounds (guarded scalar loads). LAYER0=0: K=256, branch-free b128 loads.
// ---------------------------------------------------------------------------
template <int LAYER0>
__global__ __launch_bounds__(256) void gemm_wi_kernel(
    const float* __restrict__ A, const float* __restrict__ Wk,
    float* __restrict__ wi, int K)
{
  __shared__ unsigned short Asub[128 * 40];   // 128 x 32 bf16, stride 40
  __shared__ unsigned short Bsub[64 * 40];    // [n][k] bf16, stride 40

  const int tid  = threadIdx.x;
  const int w    = tid >> 5;
  const int lane = tid & 31;
  const int q    = lane & 15;
  const int hi   = lane >> 4;
  const int m0   = blockIdx.x * 128;
  const int n0   = blockIdx.y * 64;

  v8f zf = {0.f,0.f,0.f,0.f,0.f,0.f,0.f,0.f};
  v8f acc[4] = {zf, zf, zf, zf};

  const int nk = (K + 31) >> 5;
  for (int ks = 0; ks < nk; ++ks) {
    if (LAYER0) {
      // guarded scalar path (transpose indexing, K=75 zero-pad)
      for (int idx = tid; idx < 128 * 32; idx += 256) {
        int kk = idx & 31, m = idx >> 5;
        int kg = ks * 32 + kk;
        float a = 0.f;
        if (kg < K) {
          int mg = m0 + m;
          int t = mg >> 7, b = mg & 127;
          a = A[(b * T_ + t) * I_ + kg];
        }
        Asub[m * 40 + kk] = f2bf(a);
      }
      for (int idx = tid; idx < 64 * 32; idx += 256) {
        int n = idx & 63, kk = idx >> 6;
        int kg = ks * 32 + kk;
        float b = (kg < K) ? Wk[(size_t)kg * G4_ + n0 + n] : 0.f;
        Bsub[n * 40 + kk] = f2bf(b);
      }
    } else {
      // fast path: K == 256, vectorized b128 global loads, packed LDS stores
      for (int idx = tid; idx < 128 * 8; idx += 256) {
        int m = idx >> 3, c4 = idx & 7;
        const float4 a4 = *(const float4*)&A[(size_t)(m0 + m) * H_ + ks * 32 + c4 * 4];
        uint2 p;
        p.x = pack2bf(a4.x, a4.y);
        p.y = pack2bf(a4.z, a4.w);
        *(uint2*)&Asub[m * 40 + c4 * 4] = p;
      }
      for (int idx = tid; idx < 32 * 16; idx += 256) {
        int kk = idx >> 4, n4 = idx & 15;
        const float4 b4 = *(const float4*)&Wk[(size_t)(ks * 32 + kk) * G4_ + n0 + n4 * 4];
        int n = n4 * 4;
        Bsub[(n    ) * 40 + kk] = f2bf(b4.x);
        Bsub[(n + 1) * 40 + kk] = f2bf(b4.y);
        Bsub[(n + 2) * 40 + kk] = f2bf(b4.z);
        Bsub[(n + 3) * 40 + kk] = f2bf(b4.w);
      }
    }
    __syncthreads();

    FragBF af;
    const int arow = w * 16 + q;
    const int ak0  = hi ? 8 : 0;
    af.q[0] = *(const uint4*)&Asub[arow * 40 + ak0];
    af.q[1] = *(const uint4*)&Asub[arow * 40 + ak0 + 16];
    const int bk0 = hi ? 8 : 0;
    #pragma unroll
    for (int nt = 0; nt < 4; ++nt) {
      FragBF bf_;
      int bn = nt * 16 + q;
      bf_.q[0] = *(const uint4*)&Bsub[bn * 40 + bk0];
      bf_.q[1] = *(const uint4*)&Bsub[bn * 40 + bk0 + 16];
      acc[nt] = __builtin_amdgcn_wmma_f32_16x16x32_bf16(
          false, af.v, false, bf_.v, (short)0, acc[nt], false, false);
    }
    __syncthreads();
  }

  const int r0 = m0 + w * 16 + (hi ? 8 : 0);
  #pragma unroll
  for (int nt = 0; nt < 4; ++nt) {
    int col = n0 + nt * 16 + q;
    #pragma unroll
    for (int v = 0; v < 8; ++v)
      wi[(size_t)(r0 + v) * G4_ + col] = acc[nt][v];
  }
}

// ---------------------------------------------------------------------------
// BN stats of wi over batch, per (t, column): mean and rsqrt(var+eps)
// block handles (t, 64 columns); 4 partial sums per column, fixed-order reduce.
// ---------------------------------------------------------------------------
__global__ __launch_bounds__(256) void bnstats_kernel(
    const float* __restrict__ wi, float* __restrict__ meanw, float* __restrict__ rstdw)
{
  __shared__ float s1[256], s2[256];
  const int t = blockIdx.x, n0 = blockIdx.y * 64;
  const int col = threadIdx.x & 63, part = threadIdx.x >> 6;
  float su = 0.f, sq = 0.f;
  for (int b = part * 32; b < part * 32 + 32; ++b) {
    float v = wi[(size_t)(t * B_ + b) * G4_ + n0 + col];
    su += v; sq += v * v;
  }
  s1[threadIdx.x] = su; s2[threadIdx.x] = sq;
  __syncthreads();
  if (threadIdx.x < 64) {
    su = s1[col] + s1[64 + col] + s1[128 + col] + s1[192 + col];
    sq = s2[col] + s2[64 + col] + s2[128 + col] + s2[192 + col];
    float m  = su * (1.f / 128.f);
    float vr = sq * (1.f / 128.f) - m * m;
    meanw[(size_t)t * G4_ + n0 + col] = m;
    rstdw[(size_t)t * G4_ + n0 + col] = rsqrtf(vr + EPSF);
  }
}

// ---------------------------------------------------------------------------
// Recurrent scan: 16 cooperative workgroups, WG wg owns h-channels
// [wg*16, wg*16+16) and gate columns {j, 256+j, 512+j, 768+j}.
// Per step: 128x64 GEMM (K=256) via bf16 WMMA; all BN reductions WG-local;
// cell state c lives in registers; one device barrier per step (ping-pong h).
// ---------------------------------------------------------------------------
__global__ __launch_bounds__(256) void scan_kernel(
    const float* __restrict__ Whh, const float* __restrict__ wi,
    const float* __restrict__ meanw, const float* __restrict__ rstdw,
    const float* __restrict__ bias, const float* __restrict__ g_ih,
    const float* __restrict__ b_ih, const float* __restrict__ g_hh,
    const float* __restrict__ b_hh, const float* __restrict__ g_c,
    const float* __restrict__ b_c,
    unsigned short* __restrict__ hping0, unsigned short* __restrict__ hping1,
    float* __restrict__ xout, unsigned* __restrict__ bcnt)
{
  __shared__ unsigned short WhhT[64 * 264];   // [n][k] bf16, layer-resident
  __shared__ unsigned short htile[128 * 64];  // 128 x 64 bf16 K-slice
  __shared__ float scale2[64], shift2[64], scaleH[64], shiftH[64];
  __shared__ float wps[8 * 64], wpq[8 * 64];  // per-wave column partials
  __shared__ float cps[8 * 16], cpq[8 * 16];
  __shared__ float cscale[16], cshift[16];

  const int tid   = threadIdx.x;
  const int w     = tid >> 5;
  const int lane  = tid & 31;
  const int q     = lane & 15;
  const int hi    = lane >> 4;
  const int wg    = blockIdx.x;
  const int jbase = wg * 16;
  const int jj    = jbase + q;

  // stage this WG's 64 W_hh columns, transposed, bf16 (once per layer)
  for (int idx = tid; idx < 64 * 256; idx += 256) {
    int n = idx & 63, k = idx >> 6;
    int col = (n >> 4) * 256 + jbase + (n & 15);
    WhhT[n * 264 + k] = f2bf(Whh[(size_t)k * G4_ + col]);
  }
  __syncthreads();

  float creg[8];
  #pragma unroll
  for (int v = 0; v < 8; ++v) creg[v] = 0.f;

  unsigned epoch = 0;
  v8f zf = {0.f,0.f,0.f,0.f,0.f,0.f,0.f,0.f};

  for (int t = 0; t < T_; ++t) {
    const unsigned short* hin  = (t & 1) ? hping1 : hping0;
    unsigned short*       hout = (t & 1) ? hping0 : hping1;

    // per-step affine for input-BN (scale2/shift2 fold g_ih, b_ih, bias)
    if (tid < 64) {
      int bigcol = (tid >> 4) * 256 + jbase + (tid & 15);
      float s2 = rstdw[(size_t)t * G4_ + bigcol] * g_ih[bigcol];
      scale2[tid] = s2;
      shift2[tid] = b_ih[bigcol] - meanw[(size_t)t * G4_ + bigcol] * s2 + bias[bigcol];
    }

    // --- recurrent GEMM: wh = h_prev @ Whh_cols ---
    v8f acc[4] = {zf, zf, zf, zf};
    for (int kb = 0; kb < 4; ++kb) {
      for (int idx = tid; idx < 128 * 8; idx += 256) {
        int r = idx >> 3, cq = idx & 7;
        *(uint4*)&htile[r * 64 + cq * 8] =
            *(const uint4*)&hin[r * H_ + kb * 64 + cq * 8];
      }
      __syncthreads();
      #pragma unroll
      for (int ks = 0; ks < 2; ++ks) {
        FragBF af;
        int arow = w * 16 + q;
        int ak0  = ks * 32 + (hi ? 8 : 0);
        af.q[0] = *(const uint4*)&htile[arow * 64 + ak0];
        af.q[1] = *(const uint4*)&htile[arow * 64 + ak0 + 16];
        int bk0 = kb * 64 + ks * 32 + (hi ? 8 : 0);
        #pragma unroll
        for (int nt = 0; nt < 4; ++nt) {
          FragBF bf_;
          int bn = nt * 16 + q;
          bf_.q[0] = *(const uint4*)&WhhT[bn * 264 + bk0];
          bf_.q[1] = *(const uint4*)&WhhT[bn * 264 + bk0 + 16];
          acc[nt] = __builtin_amdgcn_wmma_f32_16x16x32_bf16(
              false, af.v, false, bf_.v, (short)0, acc[nt], false, false);
        }
      }
      __syncthreads();
    }

    // --- BN(wh) stats, deterministic: lane partials -> shfl -> LDS tree ---
    #pragma unroll
    for (int nt = 0; nt < 4; ++nt) {
      float su = 0.f, sq = 0.f;
      #pragma unroll
      for (int v = 0; v < 8; ++v) { float x = acc[nt][v]; su += x; sq += x * x; }
      su += __shfl_down(su, 16);
      sq += __shfl_down(sq, 16);
      if (lane < 16) { wps[w * 64 + nt * 16 + lane] = su; wpq[w * 64 + nt * 16 + lane] = sq; }
    }
    __syncthreads();
    if (tid < 64) {
      float su = 0.f, sq = 0.f;
      #pragma unroll
      for (int ww = 0; ww < 8; ++ww) { su += wps[ww * 64 + tid]; sq += wpq[ww * 64 + tid]; }
      float m  = su * (1.f / 128.f);
      float vr = sq * (1.f / 128.f) - m * m;
      float rs = rsqrtf(vr + EPSF);
      int bigcol = (tid >> 4) * 256 + jbase + (tid & 15);
      float sc = g_hh[bigcol] * rs;
      scaleH[tid] = sc;
      shiftH[tid] = b_hh[bigcol] - m * sc;
    }
    __syncthreads();

    // --- gates + cell update (c in registers) ---
    float cn[8], ov[8];
    float csu = 0.f, csq = 0.f;
    #pragma unroll
    for (int v = 0; v < 8; ++v) {
      int b = w * 16 + hi * 8 + v;
      const float* wirow = &wi[(size_t)(t * B_ + b) * G4_];
      float s[4];
      #pragma unroll
      for (int nt = 0; nt < 4; ++nt) {
        int c = nt * 16 + q;
        s[nt] = acc[nt][v] * scaleH[c] + shiftH[c]
              + wirow[nt * 256 + jj] * scale2[c] + shift2[c];
      }
      float fg = sigm(s[0]);
      float ig = sigm(s[1]);
      float og = sigm(s[2]);
      float gg = tanhf(s[3]);
      float cnew = fg * creg[v] + ig * gg;
      creg[v] = cnew;
      cn[v] = cnew; ov[v] = og;
      csu += cnew; csq += cnew * cnew;
    }
    csu += __shfl_down(csu, 16);
    csq += __shfl_down(csq, 16);
    if (lane < 16) { cps[w * 16 + lane] = csu; cpq[w * 16 + lane] = csq; }
    __syncthreads();
    if (tid < 16) {
      float su = 0.f, sq = 0.f;
      #pragma unroll
      for (int ww = 0; ww < 8; ++ww) { su += cps[ww * 16 + tid]; sq += cpq[ww * 16 + tid]; }
      float m  = su * (1.f / 128.f);
      float vr = sq * (1.f / 128.f) - m * m;
      float rs = rsqrtf(vr + EPSF);
      float sc = g_c[jbase + tid] * rs;
      cscale[tid] = sc;
      cshift[tid] = b_c[jbase + tid] - m * sc;
    }
    __syncthreads();

    #pragma unroll
    for (int v = 0; v < 8; ++v) {
      int b = w * 16 + hi * 8 + v;
      float h = ov[v] * tanhf(cn[v] * cscale[q] + cshift[q]);
      hout[b * H_ + jj] = f2bf(h);
      xout[(size_t)(t * B_ + b) * H_ + jj] = h;
    }

    // --- device-scope barrier (epoch counter; counter memset per launch) ---
    __syncthreads();
    ++epoch;
    if (tid == 0) {
      __threadfence();
      __hip_atomic_fetch_add(bcnt, 1u, __ATOMIC_ACQ_REL, __HIP_MEMORY_SCOPE_AGENT);
      while (__hip_atomic_load(bcnt, __ATOMIC_ACQUIRE, __HIP_MEMORY_SCOPE_AGENT) < 16u * epoch)
        __builtin_amdgcn_s_sleep(1);
    }
    __syncthreads();
  }
}

// ---------------------------------------------------------------------------
// Head: out = softmax(x_last @ W_lin + b_lin), one block per batch row.
// (41 MFLOP total — negligible; scalar fp32 is fine.)
// ---------------------------------------------------------------------------
__global__ __launch_bounds__(256) void head_kernel(
    const float* __restrict__ xlast, const float* __restrict__ Wlin,
    const float* __restrict__ blin, float* __restrict__ out)
{
  __shared__ float xrow[H_];
  __shared__ float logits[C_];
  __shared__ float red[256];
  const int b = blockIdx.x;
  for (int k = threadIdx.x; k < H_; k += 256) xrow[k] = xlast[b * H_ + k];
  __syncthreads();
  for (int c = threadIdx.x; c < C_; c += 256) {
    float s = blin[c];
    for (int k = 0; k < H_; ++k) s += xrow[k] * Wlin[(size_t)k * C_ + c];
    logits[c] = s;
  }
  __syncthreads();
  float mx = -3.0e38f;
  for (int c = threadIdx.x; c < C_; c += 256) mx = fmaxf(mx, logits[c]);
  red[threadIdx.x] = mx; __syncthreads();
  for (int s = 128; s > 0; s >>= 1) {
    if (threadIdx.x < s) red[threadIdx.x] = fmaxf(red[threadIdx.x], red[threadIdx.x + s]);
    __syncthreads();
  }
  mx = red[0]; __syncthreads();
  float se = 0.f;
  for (int c = threadIdx.x; c < C_; c += 256) se += __expf(logits[c] - mx);
  red[threadIdx.x] = se; __syncthreads();
  for (int s = 128; s > 0; s >>= 1) {
    if (threadIdx.x < s) red[threadIdx.x] += red[threadIdx.x + s];
    __syncthreads();
  }
  float inv = 1.f / red[0];
  for (int c = threadIdx.x; c < C_; c += 256) out[(size_t)b * C_ + c] = __expf(logits[c] - mx) * inv;
}

// ---------------------------------------------------------------------------
extern "C" void kernel_launch(void* const* d_in, const int* in_sizes, int n_in,
                              void* d_out, int out_size, void* d_ws, size_t ws_size,
                              hipStream_t stream) {
  (void)in_sizes; (void)n_in; (void)out_size;
  const float* seq   = (const float*)d_in[0];
  const float* W_ih0 = (const float*)d_in[1];
  const float* W_ih  = (const float*)d_in[2];
  const float* W_hh  = (const float*)d_in[3];
  const float* bias  = (const float*)d_in[4];
  const float* g_ih  = (const float*)d_in[5];
  const float* b_ih  = (const float*)d_in[6];
  const float* g_hh  = (const float*)d_in[7];
  const float* b_hh  = (const float*)d_in[8];
  const float* g_c   = (const float*)d_in[9];
  const float* b_c   = (const float*)d_in[10];
  const float* W_lin = (const float*)d_in[11];
  const float* b_lin = (const float*)d_in[12];
  float* out = (float*)d_out;

  char* ws = (char*)d_ws;
  size_t off = 0;
  auto alloc = [&](size_t bytes) { void* p = ws + off; off = (off + bytes + 255) & ~(size_t)255; return p; };
  float*          wi     = (float*)alloc((size_t)T_ * B_ * G4_ * 4);
  float*          meanw  = (float*)alloc((size_t)T_ * G4_ * 4);
  float*          rstdw  = (float*)alloc((size_t)T_ * G4_ * 4);
  float*          x      = (float*)alloc((size_t)T_ * B_ * H_ * 4);
  unsigned short* hping0 = (unsigned short*)alloc((size_t)B_ * H_ * 2);
  unsigned short* hping1 = (unsigned short*)alloc((size_t)B_ * H_ * 2);
  unsigned*       bcnt   = (unsigned*)alloc(256);
  if (off > ws_size) return;  // workspace too small; avoid OOB

  for (int l = 0; l < L_; ++l) {
    const float* Wk = (l == 0) ? W_ih0 : (W_ih + (size_t)(l - 1) * H_ * G4_);
    if (l == 0)
      gemm_wi_kernel<1><<<dim3(T_, 16), 256, 0, stream>>>(seq, Wk, wi, I_);
    else
      gemm_wi_kernel<0><<<dim3(T_, 16), 256, 0, stream>>>(x, Wk, wi, H_);
    bnstats_kernel<<<dim3(T_, 16), 256, 0, stream>>>(wi, meanw, rstdw);
    hipMemsetAsync(hping0, 0, (size_t)B_ * H_ * 2, stream);
    hipMemsetAsync(bcnt, 0, sizeof(unsigned), stream);
    scan_kernel<<<16, 256, 0, stream>>>(
        W_hh + (size_t)l * H_ * G4_, wi, meanw, rstdw,
        bias + (size_t)l * G4_, g_ih + (size_t)l * G4_, b_ih + (size_t)l * G4_,
        g_hh + (size_t)l * G4_, b_hh + (size_t)l * G4_,
        g_c + (size_t)l * H_, b_c + (size_t)l * H_,
        hping0, hping1, x, bcnt);
  }
  head_kernel<<<B_, 256, 0, stream>>>(x + (size_t)(T_ - 1) * B_ * H_, W_lin, b_lin, out);
}
